// VQAModel_Attention_75892072120722
// MI455X (gfx1250) — matrix operand order repaired and checked by reading, not compiled, not run on hardware
//
#include <hip/hip_runtime.h>

// ---------------------------------------------------------------------------
// VQA attention model forward for MI455X (gfx1250), wave32.
// Dense GEMMs: v_wmma_f32_16x16x32_f16, 4 WMMAs/wave/K-step (16x64 strip),
//   A-tile staged via global_load_async_to_lds_b128 (ASYNCcnt),
//   B-tile transposed into LDS by VALU.
// Convs: implicit-im2col WMMA GEMM, 2 WMMAs/wave/K-step (32x16 strip).
// ---------------------------------------------------------------------------

typedef _Float16 half8  __attribute__((ext_vector_type(8)));
typedef _Float16 half16 __attribute__((ext_vector_type(16)));
typedef float    f32x8  __attribute__((ext_vector_type(8)));

#define DEV static __device__ __forceinline__

#define BQ    64
#define TQn   20
#define TAn   20
#define HIDn  512
#define EMBn  300
#define FDIMn 256
#define Vn    30000
#define KQP   832     // 300+512 padded to 32
#define KDP   1088    // 300+256+512 padded to 32

// ---------------- fragments (CDNA5 ISA 7.12.2 layouts) ---------------------
DEV half16 frag_a_lds(const _Float16* lA /*16x32 row-major*/) {
    int lane = threadIdx.x & 31;
    int r    = lane & 15;
    int koff = (lane >> 4) * 8;      // lanes 0-15: K 0-7/16-23; lanes 16-31: K 8-15/24-31
    half8 lo = *(const half8*)(lA + r * 32 + koff);
    half8 hi = *(const half8*)(lA + r * 32 + 16 + koff);
    half16 a;
#pragma unroll
    for (int i = 0; i < 8; ++i) { a[i] = lo[i]; a[8 + i] = hi[i]; }
    return a;
}

DEV half16 frag_b_lds(const _Float16* lB /*32 x stride, [k][n]*/, int stride, int ncol0) {
    int lane = threadIdx.x & 31;     // lane%16 = N col; lanes 0-15: K 0-7/16-23; lanes 16-31: K 8-15/24-31
    int n    = lane & 15;
    int koff = (lane >> 4) * 8;
    half16 b;
#pragma unroll
    for (int i = 0; i < 8; ++i) {
        b[i]     = lB[(koff + i) * stride + ncol0 + n];
        b[8 + i] = lB[(16 + koff + i) * stride + ncol0 + n];
    }
    return b;
}

// ---------------- dense GEMM: C[M,N] = A[M,K] * W[N,ldw]^T + bias ----------
// Requires K % 32 == 0 and M % 32 == 0 (buffers padded accordingly).
// Block tile 32M x 256N, 8 waves: wave (mi=wv>>2, ni=wv&3) -> 16M x 64N strip.
__global__ void k_gemm_f16(const _Float16* __restrict__ A,
                           const _Float16* __restrict__ W,
                           const float* __restrict__ bias,
                           float* __restrict__ C,
                           int M, int N, int K, int ldw, int ldc) {
    __shared__ __align__(16) _Float16 lA[32 * 32];
    __shared__ __align__(16) _Float16 lB[32 * 256];
    const int m0   = blockIdx.x * 32;
    const int n0b  = blockIdx.y * 256;
    const int wv   = threadIdx.x >> 5;
    const int lane = threadIdx.x & 31;
    const int mi   = wv >> 2, ni = wv & 3;
    const int n0w  = n0b + ni * 64;
    f32x8 acc[4] = {{}, {}, {}, {}};
    const int KT = K >> 5;
    for (int kt = 0; kt < KT; ++kt) {
        const int k0 = kt * 32;
        if (wv == 0) {
            // async DMA: 32 rows x 64B of A into LDS (4 x wave-wide B128)
            int r0 = lane >> 2;          // 0..7
            int ch = lane & 3;           // 16B chunk within row
            const _Float16* gp = A + (size_t)(m0 + r0) * K + k0 + ch * 8;
            unsigned lp = (unsigned)(size_t)(lA + r0 * 32 + ch * 8);
#pragma unroll
            for (int i = 0; i < 4; ++i) {
                asm volatile("global_load_async_to_lds_b128 %0, %1, off"
                             :: "v"(lp + (unsigned)(i * 8 * 64)),
                                "v"((unsigned long long)(gp + (size_t)i * 8 * K))
                             : "memory");
            }
        }
        {   // B staging: transpose W rows into lB[k][n], vectorized global reads
            int n = n0b + threadIdx.x;
            if (n < N) {
#pragma unroll
                for (int j = 0; j < 4; ++j) {
                    half8 w = *(const half8*)&W[(size_t)n * ldw + k0 + j * 8];
#pragma unroll
                    for (int e = 0; e < 8; ++e)
                        lB[(j * 8 + e) * 256 + threadIdx.x] = w[e];
                }
            } else {
#pragma unroll
                for (int j = 0; j < 32; ++j) lB[j * 256 + threadIdx.x] = (_Float16)0.f;
            }
        }
        if (wv == 0) asm volatile("s_wait_asynccnt 0x0" ::: "memory");
        __syncthreads();
        {
            half16 a = frag_a_lds(lA + mi * 16 * 32);
#pragma unroll
            for (int s = 0; s < 4; ++s) {
                half16 b = frag_b_lds(lB, 256, ni * 64 + s * 16);
                acc[s] = __builtin_amdgcn_wmma_f32_16x16x32_f16(false, a, false, b,
                                                                (short)0, acc[s], false, false);
            }
        }
        __syncthreads();
    }
    {
        int nl = lane & 15, mb = (lane >> 4) * 8;
#pragma unroll
        for (int s = 0; s < 4; ++s) {
            int n = n0w + s * 16 + nl;
            if (n < N) {
                float bv = bias ? bias[n] : 0.f;
#pragma unroll
                for (int r = 0; r < 8; ++r)
                    C[(size_t)(m0 + mi * 16 + mb + r) * ldc + n] = acc[s][r] + bv;
            }
        }
    }
}

// ------------- implicit-im2col conv GEMM (3x3 SAME) + bias + ReLU ----------
// rows m = (b,y,x), K = Cin*9 (guarded).  Block tile 128M x 32N, 8 waves:
// wave (mi=wv>>1, ni=wv&1) -> 32M x 16N strip (2 WMMAs sharing the B frag).
__global__ void k_conv_gemm(const float* __restrict__ in,
                            const float* __restrict__ wc,
                            const float* __restrict__ bias,
                            float* __restrict__ out,
                            int Cin, int H, int Wd, int Cout) {
    __shared__ __align__(16) _Float16 lA[128 * 32];
    __shared__ __align__(16) _Float16 lB[32 * 32];
    const int HW   = H * Wd;
    const int m0   = blockIdx.x * 128;
    const int n0b  = blockIdx.y * 32;
    const int wv   = threadIdx.x >> 5;
    const int lane = threadIdx.x & 31;
    const int mi   = wv >> 1, ni = wv & 1;
    const int n0w  = n0b + ni * 16;
    const int K    = Cin * 9;
    f32x8 acc0 = {}, acc1 = {};
    const int KT = (K + 31) >> 5;
    for (int kt = 0; kt < KT; ++kt) {
        const int k0 = kt * 32;
        {   // implicit im2col A staging: 128 rows x 32 k, 16 elems/thread
            int r  = threadIdx.x >> 1;
            int kh = (threadIdx.x & 1) * 16;
            int m = m0 + r;
            int b = m / HW; int rem = m - b * HW;
            int y = rem / Wd; int x = rem - y * Wd;
#pragma unroll
            for (int j = 0; j < 16; ++j) {
                int k = k0 + kh + j;
                float v = 0.f;
                if (k < K) {
                    int c = k / 9; int r9 = k - c * 9;
                    int dy = r9 / 3 - 1, dx = r9 % 3 - 1;
                    int yy = y + dy, xx = x + dx;
                    if (yy >= 0 && yy < H && xx >= 0 && xx < Wd)
                        v = in[(size_t)((b * Cin + c) * H + yy) * Wd + xx];
                }
                lA[r * 32 + kh + j] = (_Float16)v;
            }
        }
        {   // weight staging f32->f16 transposed into lB[k][n], 4 elems/thread
            int nl = threadIdx.x & 31;
            int kq = (threadIdx.x >> 5) * 4;
            int n  = n0b + nl;
#pragma unroll
            for (int j = 0; j < 4; ++j) {
                int k = k0 + kq + j;
                _Float16 v = (_Float16)0.f;
                if (n < Cout && k < K) v = (_Float16)wc[(size_t)n * K + k];
                lB[(kq + j) * 32 + nl] = v;
            }
        }
        __syncthreads();
        if (n0w < Cout) {
            half16 b  = frag_b_lds(lB, 32, ni * 16);
            half16 a0 = frag_a_lds(lA + (mi * 32) * 32);
            half16 a1 = frag_a_lds(lA + (mi * 32 + 16) * 32);
            acc0 = __builtin_amdgcn_wmma_f32_16x16x32_f16(false, a0, false, b,
                                                          (short)0, acc0, false, false);
            acc1 = __builtin_amdgcn_wmma_f32_16x16x32_f16(false, a1, false, b,
                                                          (short)0, acc1, false, false);
        }
        __syncthreads();
    }
    if (n0w < Cout) {
        int nl = lane & 15, mb = (lane >> 4) * 8;
        int n = n0w + nl;
        float bv = bias[n];
#pragma unroll
        for (int f = 0; f < 2; ++f) {
#pragma unroll
            for (int r = 0; r < 8; ++r) {
                int m = m0 + mi * 32 + f * 16 + mb + r;
                int b = m / HW; int rem = m - b * HW;
                int y = rem / Wd; int x = rem - y * Wd;
                float v = (f ? acc1[r] : acc0[r]) + bv;
                v = v > 0.f ? v : 0.f;
                out[(size_t)((b * Cout + n) * H + y) * Wd + x] = v;
            }
        }
    }
}

// ---------------- small elementwise / reduction kernels --------------------
__global__ void k_pool2x2(const float* __restrict__ in, float* __restrict__ out,
                          int total, int C, int Ho, int Wo) {
    int i = blockIdx.x * blockDim.x + threadIdx.x;
    if (i >= total) return;
    int wo = i % Wo; int t = i / Wo; int ho = t % Ho; t /= Ho;
    int c = t % C; int b = t / C;
    int W2 = Wo * 2;
    const float* p = in + (size_t)((b * C + c) * (Ho * 2) + ho * 2) * W2 + wo * 2;
    out[i] = fmaxf(fmaxf(p[0], p[1]), fmaxf(p[W2], p[W2 + 1]));
}

__global__ void k_cvt_f16(const float* __restrict__ s, _Float16* __restrict__ d, int n) {
    int i = blockIdx.x * blockDim.x + threadIdx.x;
    if (i < n) d[i] = (_Float16)s[i];
}

// dst f16 [N, Kp]: row n = [w1 row n | w2 row n | 0-pad]
__global__ void k_build_cat(_Float16* __restrict__ dst,
                            const float* __restrict__ w1, int k1,
                            const float* __restrict__ w2, int k2, int N, int Kp) {
    int i = blockIdx.x * blockDim.x + threadIdx.x;
    if (i >= N * Kp) return;
    int n = i / Kp, k = i - n * Kp;
    float v = 0.f;
    if (k < k1)            v = w1[(size_t)n * k1 + k];
    else if (k < k1 + k2)  v = w2[(size_t)n * k2 + (k - k1)];
    dst[i] = (_Float16)v;
}

__global__ void k_add2(float* __restrict__ d, const float* __restrict__ a,
                       const float* __restrict__ b, int n) {
    int i = blockIdx.x * blockDim.x + threadIdx.x;
    if (i < n) d[i] = a[i] + b[i];
}

__global__ void k_zero_f32(float* __restrict__ p, int n) {
    int i = blockIdx.x * blockDim.x + threadIdx.x;
    if (i < n) p[i] = 0.f;
}

// featAin[(b*64+n)*256 + f] = p4[b][c=f][y*8+x=n]
__global__ void k_featain(const float* __restrict__ p4, _Float16* __restrict__ fa) {
    int i = blockIdx.x * blockDim.x + threadIdx.x;
    if (i >= 4096 * 256) return;
    int m = i >> 8, f = i & 255;
    int b = m >> 6, n = m & 63;
    fa[i] = (_Float16)p4[(size_t)((b * 256 + f) * 64) + n];
}

// xh[b][k]: k<300 -> emb[tok[b][t]]; k<300+Kctx -> ctx; k<valid -> h; else 0
__global__ void k_prep_xh(_Float16* __restrict__ xh, const float* __restrict__ emb,
                          const int* __restrict__ tok, int T, int t,
                          const float* __restrict__ ctx, int Kctx,
                          const float* __restrict__ h, int Kp) {
    int i = blockIdx.x * blockDim.x + threadIdx.x;
    if (i >= BQ * Kp) return;
    int b = i / Kp, k = i - b * Kp;
    int valid = EMBn + Kctx + HIDn;
    float v = 0.f;
    if (k < EMBn)               v = emb[(size_t)tok[b * T + t] * EMBn + k];
    else if (k < EMBn + Kctx)   v = ctx[b * Kctx + (k - EMBn)];
    else if (k < valid)         v = h[b * HIDn + (k - EMBn - Kctx)];
    xh[i] = (_Float16)v;
}

// gates [B,2048] (i,f,g,o) -> update h,c; emit h16 for the next WMMA GEMMs
__global__ void k_lstm_pw(const float* __restrict__ gates, float* __restrict__ h,
                          float* __restrict__ c, _Float16* __restrict__ h16) {
    int i = blockIdx.x * blockDim.x + threadIdx.x;
    if (i >= BQ * HIDn) return;
    int b = i >> 9, j = i & (HIDn - 1);
    const float* g = gates + (size_t)b * 4 * HIDn;
    float gi = g[j], gf = g[HIDn + j], gg = g[2 * HIDn + j], go = g[3 * HIDn + j];
    float si = 1.f / (1.f + __expf(-gi));
    float sf = 1.f / (1.f + __expf(-gf));
    float so = 1.f / (1.f + __expf(-go));
    float cn = sf * c[i] + si * tanhf(gg);
    float hn = so * tanhf(cn);
    c[i] = cn; h[i] = hn; h16[i] = (_Float16)hn;
}

// e[b*64+n] = sum_j tanh(featA[row][j] + hA[b][j]) * vw[j]   (wave per row)
__global__ void k_energy(const float* __restrict__ featA, const float* __restrict__ hA,
                         const float* __restrict__ vw, float* __restrict__ e) {
    int row = blockIdx.x * 8 + (threadIdx.x >> 5);
    int lane = threadIdx.x & 31;
    int b = row >> 6;
    float acc = 0.f;
    for (int j = lane; j < HIDn; j += 32)
        acc += tanhf(featA[(size_t)row * HIDn + j] + hA[b * HIDn + j]) * vw[j];
#pragma unroll
    for (int o = 16; o > 0; o >>= 1) acc += __shfl_down(acc, o, 32);
    if (lane == 0) e[row] = acc;
}

// per batch b: softmax over 64 energies, ctx[b][f] = sum_n alpha[n]*feat[b][n][f]
__global__ void k_softmax_ctx(const float* __restrict__ e, const float* __restrict__ p4,
                              float* __restrict__ ctx) {
    __shared__ float sa[64];
    int b = blockIdx.x, tid = threadIdx.x;
    if (tid < 64) sa[tid] = e[b * 64 + tid];
    __syncthreads();
    if (tid == 0) {
        float mx = sa[0];
        for (int n = 1; n < 64; ++n) mx = fmaxf(mx, sa[n]);
        float s = 0.f;
        for (int n = 0; n < 64; ++n) { sa[n] = __expf(sa[n] - mx); s += sa[n]; }
        float inv = 1.f / s;
        for (int n = 0; n < 64; ++n) sa[n] *= inv;
    }
    __syncthreads();
    int f = tid;                    // 256 threads == FDIM
    float acc = 0.f;
    const float* fp = p4 + (size_t)(b * 256 + f) * 64;
    for (int n = 0; n < 64; ++n) acc += sa[n] * fp[n];
    ctx[b * 256 + f] = acc;
}

// ---------------- workspace layout (bytes) ---------------------------------
static constexpr size_t alg(size_t x) { return (x + 255) & ~(size_t)255; }
static constexpr size_t O_BIG = 0;                                    // 64*32*128*128*4
static constexpr size_t O_P1  = alg(O_BIG + 134217728);               // 64*32*64*64*4
static constexpr size_t O_P2  = alg(O_P1 + 33554432);                 // 64*64*32*32*4
static constexpr size_t O_P3  = alg(O_P2 + 16777216);                 // 64*128*16*16*4
static constexpr size_t O_P4  = alg(O_P3 + 8388608);                  // 64*256*8*8*4
static constexpr size_t O_FCW = alg(O_P4 + 4194304);                  // 30000*512*2
static constexpr size_t O_QW  = alg(O_FCW + (size_t)Vn * HIDn * 2);   // 2048*832*2
static constexpr size_t O_DW  = alg(O_QW + 2048 * KQP * 2);           // 2048*1088*2
static constexpr size_t O_AT  = alg(O_DW + 2048 * KDP * 2);           // 512*768*2
static constexpr size_t O_FAI = alg(O_AT + 512 * 768 * 2);            // 4096*256*2
static constexpr size_t O_FA  = alg(O_FAI + 4096 * 256 * 2);          // 4096*512*4
static constexpr size_t O_H   = alg(O_FA + 4096 * 512 * 4);           // 64*512*4
static constexpr size_t O_C   = alg(O_H + 131072);
static constexpr size_t O_H16 = alg(O_C + 131072);                    // 64*512*2
static constexpr size_t O_GT  = alg(O_H16 + 65536);                   // 64*2048*4
static constexpr size_t O_HA  = alg(O_GT + 524288);                   // 64*512*4
static constexpr size_t O_E   = alg(O_HA + 131072);                   // 4096*4
static constexpr size_t O_CTX = alg(O_E + 16384);                     // 64*256*4
static constexpr size_t O_QB  = alg(O_CTX + 65536);                   // 2048*4
static constexpr size_t O_DB  = alg(O_QB + 8192);
static constexpr size_t O_XHQ = alg(O_DB + 8192);                     // 64*832*2
static constexpr size_t O_XHD = alg(O_XHQ + (size_t)BQ * KQP * 2);    // 64*1088*2

extern "C" void kernel_launch(void* const* d_in, const int* in_sizes, int n_in,
                              void* d_out, int out_size, void* d_ws, size_t ws_size,
                              hipStream_t stream) {
    (void)in_sizes; (void)n_in; (void)out_size; (void)ws_size;
    char* ws = (char*)d_ws;
    const float* img    = (const float*)d_in[0];
    const int*   ques   = (const int*)d_in[1];
    const int*   ans    = (const int*)d_in[2];
    const float* c1w = (const float*)d_in[3],  * c1b = (const float*)d_in[4];
    const float* c2w = (const float*)d_in[5],  * c2b = (const float*)d_in[6];
    const float* c3w = (const float*)d_in[7],  * c3b = (const float*)d_in[8];
    const float* c4w = (const float*)d_in[9],  * c4b = (const float*)d_in[10];
    const float* q_emb = (const float*)d_in[11];
    const float* qWih = (const float*)d_in[12], * qWhh = (const float*)d_in[13];
    const float* qbih = (const float*)d_in[14], * qbhh = (const float*)d_in[15];
    const float* attn_w = (const float*)d_in[16], * attn_b = (const float*)d_in[17];
    const float* v_w = (const float*)d_in[18];
    const float* d_emb = (const float*)d_in[19];
    const float* dWih = (const float*)d_in[20], * dWhh = (const float*)d_in[21];
    const float* dbih = (const float*)d_in[22], * dbhh = (const float*)d_in[23];
    const float* fc_w = (const float*)d_in[24], * fc_b = (const float*)d_in[25];
    float* out = (float*)d_out;

    float*    BIG = (float*)(ws + O_BIG);
    float*    P1  = (float*)(ws + O_P1);
    float*    P2  = (float*)(ws + O_P2);
    float*    P3  = (float*)(ws + O_P3);
    float*    P4  = (float*)(ws + O_P4);
    _Float16* FCW = (_Float16*)(ws + O_FCW);
    _Float16* QW  = (_Float16*)(ws + O_QW);
    _Float16* DW  = (_Float16*)(ws + O_DW);
    _Float16* AT  = (_Float16*)(ws + O_AT);
    _Float16* FAI = (_Float16*)(ws + O_FAI);
    float*    FA  = (float*)(ws + O_FA);
    float*    H   = (float*)(ws + O_H);
    float*    Cst = (float*)(ws + O_C);
    _Float16* H16 = (_Float16*)(ws + O_H16);
    float*    GT  = (float*)(ws + O_GT);
    float*    HA  = (float*)(ws + O_HA);
    float*    E   = (float*)(ws + O_E);
    float*    CTX = (float*)(ws + O_CTX);
    float*    QB  = (float*)(ws + O_QB);
    float*    DB  = (float*)(ws + O_DB);
    _Float16* XHQ = (_Float16*)(ws + O_XHQ);
    _Float16* XHD = (_Float16*)(ws + O_XHD);

    auto g1 = [](int n) { return dim3((unsigned)((n + 255) / 256)); };

    // ---- weight conversions (idempotent per call) ----
    k_cvt_f16<<<g1(512 * 768), 256, 0, stream>>>(attn_w, AT, 512 * 768);
    k_cvt_f16<<<g1(Vn * HIDn), 256, 0, stream>>>(fc_w, FCW, Vn * HIDn);
    k_build_cat<<<g1(2048 * KQP), 256, 0, stream>>>(QW, qWih, EMBn, qWhh, HIDn, 4 * HIDn, KQP);
    k_build_cat<<<g1(2048 * KDP), 256, 0, stream>>>(DW, dWih, EMBn + FDIMn, dWhh, HIDn, 4 * HIDn, KDP);
    k_add2<<<g1(2048), 256, 0, stream>>>(QB, qbih, qbhh, 4 * HIDn);
    k_add2<<<g1(2048), 256, 0, stream>>>(DB, dbih, dbhh, 4 * HIDn);

    // ---- CNN: conv(WMMA implicit im2col) + pool, 128->64->32->16->8 ----
    k_conv_gemm<<<dim3(64 * 128 * 128 / 128, 1), 256, 0, stream>>>(img, c1w, c1b, BIG, 3, 128, 128, 32);
    k_pool2x2<<<g1(64 * 32 * 64 * 64), 256, 0, stream>>>(BIG, P1, 64 * 32 * 64 * 64, 32, 64, 64);
    k_conv_gemm<<<dim3(64 * 64 * 64 / 128, 2), 256, 0, stream>>>(P1, c2w, c2b, BIG, 32, 64, 64, 64);
    k_pool2x2<<<g1(64 * 64 * 32 * 32), 256, 0, stream>>>(BIG, P2, 64 * 64 * 32 * 32, 64, 32, 32);
    k_conv_gemm<<<dim3(64 * 32 * 32 / 128, 4), 256, 0, stream>>>(P2, c3w, c3b, BIG, 64, 32, 32, 128);
    k_pool2x2<<<g1(64 * 128 * 16 * 16), 256, 0, stream>>>(BIG, P3, 64 * 128 * 16 * 16, 128, 16, 16);
    k_conv_gemm<<<dim3(64 * 16 * 16 / 128, 8), 256, 0, stream>>>(P3, c4w, c4b, BIG, 128, 16, 16, 256);
    k_pool2x2<<<g1(64 * 256 * 8 * 8), 256, 0, stream>>>(BIG, P4, 64 * 256 * 8 * 8, 256, 8, 8);

    // ---- hoisted feature half of attention: FA = feat @ attn_w[:, :256]^T + attn_b ----
    k_featain<<<g1(4096 * 256), 256, 0, stream>>>(P4, FAI);
    k_gemm_f16<<<dim3(4096 / 32, 2), 256, 0, stream>>>(FAI, AT, attn_b, FA,
                                                       4096, HIDn, FDIMn, 768, HIDn);

    // ---- question encoder LSTM ----
    k_zero_f32<<<g1(BQ * HIDn), 256, 0, stream>>>(H, BQ * HIDn);
    k_zero_f32<<<g1(BQ * HIDn), 256, 0, stream>>>(Cst, BQ * HIDn);
    for (int t = 0; t < TQn; ++t) {
        k_prep_xh<<<g1(BQ * KQP), 256, 0, stream>>>(XHQ, q_emb, ques, TQn, t, nullptr, 0, H, KQP);
        k_gemm_f16<<<dim3(BQ / 32, 8), 256, 0, stream>>>(XHQ, QW, QB, GT,
                                                         BQ, 4 * HIDn, KQP, KQP, 4 * HIDn);
        k_lstm_pw<<<g1(BQ * HIDn), 256, 0, stream>>>(GT, H, Cst, H16);
    }

    // ---- attention decoder ----
    k_zero_f32<<<g1(BQ * HIDn), 256, 0, stream>>>(Cst, BQ * HIDn);
    for (int t = 0; t < TAn; ++t) {
        // hA = h @ attn_w[:, 256:]^T   (pre-step h via H16)
        k_gemm_f16<<<dim3(BQ / 32, 2), 256, 0, stream>>>(H16, AT + FDIMn, nullptr, HA,
                                                         BQ, HIDn, HIDn, 768, HIDn);
        k_energy<<<512, 256, 0, stream>>>(FA, HA, v_w, E);
        k_softmax_ctx<<<BQ, 256, 0, stream>>>(E, P4, CTX);
        k_prep_xh<<<g1(BQ * KDP), 256, 0, stream>>>(XHD, d_emb, ans, TAn, t, CTX, FDIMn, H, KDP);
        k_gemm_f16<<<dim3(BQ / 32, 8), 256, 0, stream>>>(XHD, DW, DB, GT,
                                                         BQ, 4 * HIDn, KDP, KDP, 4 * HIDn);
        k_lstm_pw<<<g1(BQ * HIDn), 256, 0, stream>>>(GT, H, Cst, H16);
        // logits_t = h_new @ fc_w^T + fc_b  -> out[b][t][v], ldc = TA*V
        k_gemm_f16<<<dim3(BQ / 32, (Vn + 255) / 256), 256, 0, stream>>>(
            H16, FCW, fc_b, out + (size_t)t * Vn, BQ, Vn, HIDn, HIDn, TAn * Vn);
    }
}
